// CrossAttentionLayer_1924145348697
// MI455X (gfx1250) — matrix operand order repaired and checked by reading, not compiled, not run on hardware
//
#include <hip/hip_runtime.h>

// CDNA5 / gfx1250 fused cross-attention layer.
//   Q = Wq x_slice + bq                         (scalar, tiny)
//   K = Wk x_vol + bk  -> s = q*k*scale -> LDS  (bf16 WMMA 16x16x32, fp32 accum)
//   softmax over d (256) in LDS                  (fp32)
//   V = Wv x_vol + bv  -> out = p * v           (bf16 WMMA, fp32 out)
// One workgroup = 8 hw pixels, all 64 channels, all 256 d. Logits live in
// 256 KB of LDS (possible only with CDNA5's 320 KB/WGP), so K/V/logits are
// never written to HBM: traffic ~= read x_vol (1-2x) + write out.

typedef __attribute__((ext_vector_type(16))) __bf16 v16bf;
typedef __attribute__((ext_vector_type(8)))  float  v8f;

#define C_DIM   64
#define D_DIM   256
#define HW_DIM  4096
#define HWT     8                       // hw pixels per workgroup
#define CH_VOL  (D_DIM * HW_DIM)        // channel stride in x_volume (elements)
#define SCALE   0.125f                  // 1/sqrt(C)

#define S_ELEMS  (D_DIM * C_DIM * HWT)  // 131072 bf16 logits
#define SMEM_S   (S_ELEMS * 2)          // 256 KB
#define SMEM_Q   (C_DIM * HWT * 4)      // 2 KB
#define SMEM_INV (C_DIM * HWT * 4)      // 2 KB
#define SMEM_BK  (C_DIM * 4)
#define SMEM_BV  (C_DIM * 4)
#define SMEM_BYTES (SMEM_S + SMEM_Q + SMEM_INV + SMEM_BK + SMEM_BV)  // 266752

__global__ __launch_bounds__(256)
void xattn_fused(const float* __restrict__ x_slice,
                 const float* __restrict__ x_vol,
                 const float* __restrict__ Wq, const float* __restrict__ bq,
                 const float* __restrict__ Wk, const float* __restrict__ bk,
                 const float* __restrict__ Wv, const float* __restrict__ bv,
                 float* __restrict__ out)
{
  extern __shared__ char smem[];
  __bf16* s_lds   = (__bf16*)smem;                                   // [d][o][hw]
  float*  q_lds   = (float*)(smem + SMEM_S);                         // [o][hw]
  float*  inv_lds = (float*)(smem + SMEM_S + SMEM_Q);                // [o][hw]
  float*  bk_lds  = (float*)(smem + SMEM_S + SMEM_Q + SMEM_INV);
  float*  bv_lds  = (float*)(smem + SMEM_S + SMEM_Q + SMEM_INV + SMEM_BK);

  const int tid    = threadIdx.x;
  const int lane   = tid & 31;
  const int wave   = tid >> 5;          // 8 waves
  const int hwbase = blockIdx.x * HWT;  // 512 workgroups cover 4096 pixels

  // wave32 WMMA tile decomposition (lane-constant)
  const int half   = lane >> 4;         // 0: lanes 0-15, 1: lanes 16-31
  const int nlane  = lane & 15;         // column index within 16-wide tile
  const int dl_off = nlane >> 3;        // column -> local d offset (0/1)
  const int hwl    = nlane & 7;         // column -> hw offset

  // ---------------- pass 0: stage biases, compute Q ----------------
  if (tid < C_DIM) { bk_lds[tid] = bk[tid]; bv_lds[tid] = bv[tid]; }
  for (int p = tid; p < C_DIM * HWT; p += 256) {
    const int o = p >> 3, h = p & 7;
    float acc = bq[o];
    for (int i = 0; i < C_DIM; ++i)
      acc = fmaf(Wq[o * C_DIM + i], x_slice[i * HW_DIM + hwbase + h], acc);
    q_lds[p] = acc;
  }
  __syncthreads();

  // A-tile loader: W (64x64 row-major fp32) -> bf16 WMMA A layout (16x32/tile).
  // Order used here (not ISA table order): lane M = lane%16; K = (lane/16)*16 + e (contiguous, matches B).
  v16bf A[4][2];
  auto load_A = [&](const float* __restrict__ W) {
    const int kb = half * 8;
#pragma unroll
    for (int ot = 0; ot < 4; ++ot)
#pragma unroll
      for (int kt = 0; kt < 2; ++kt) {
        v16bf a;
#pragma unroll
        for (int e = 0; e < 16; ++e) {
          const int K = kt * 32 + half * 16 + e;
          a[e] = (__bf16)W[(ot * 16 + nlane) * C_DIM + K];
        }
        A[ot][kt] = a;
      }
  };

  // One projection pass over this wave's 16 d-pair column tiles.
  // vpass=false: K projection -> logits to LDS.  vpass=true: V -> final output.
  auto gemm_pass = [&](const float* __restrict__ bias_l, bool vpass) {
    const int p0 = wave * 16;           // disjoint d-pair range per wave
    for (int p = p0; p < p0 + 16; ++p) {
      const int  dl      = 2 * p + dl_off;                 // this lane's d
      const long colAddr = (long)dl * HW_DIM + (hwbase + hwl);
      // B tiles 32x16 bf16: col N = lane%16, K(i) = kt*32 + (lane/16)*16 + e
      v16bf B[2];
#pragma unroll
      for (int kt = 0; kt < 2; ++kt) {
        const int i0 = kt * 32 + half * 16;
        v16bf b;
#pragma unroll
        for (int e = 0; e < 16; ++e)
          b[e] = (__bf16)x_vol[(long)(i0 + e) * CH_VOL + colAddr];
        B[kt] = b;
      }
      if (p + 1 < p0 + 16)  // hint next d-pair (gfx1250 global_prefetch_b8)
        __builtin_prefetch(&x_vol[(long)(half * 16) * CH_VOL + colAddr + 2 * HW_DIM], 0, 1);
#pragma unroll
      for (int ot = 0; ot < 4; ++ot) {
        v8f c = {};
        c = __builtin_amdgcn_wmma_f32_16x16x32_bf16(false, A[ot][0], false, B[0],
                                                    (short)0, c, false, false);
        c = __builtin_amdgcn_wmma_f32_16x16x32_bf16(false, A[ot][1], false, B[1],
                                                    (short)0, c, false, false);
        const int obase = ot * 16 + half * 8;   // D layout: M = (lane/16)*8 + r
#pragma unroll
        for (int r = 0; r < 8; ++r) {
          const int   o    = obase + r;
          const float proj = c[r] + bias_l[o];
          const int   sidx = (dl * C_DIM + o) * HWT + hwl;
          if (!vpass) {
            const float s = q_lds[o * HWT + hwl] * proj * SCALE;
            s_lds[sidx] = (__bf16)s;
          } else {
            const float pr = (float)s_lds[sidx] * inv_lds[o * HWT + hwl];
            out[(long)o * CH_VOL + colAddr] = pr * proj;
          }
        }
      }
    }
  };

  // ---------------- pass 1: K projection + logits ----------------
  load_A(Wk);
  gemm_pass(bk_lds, false);
  __syncthreads();

  // ---------------- pass 2: softmax over d (axis 2) ----------------
  for (int p = tid; p < C_DIM * HWT; p += 256) {
    __bf16* col = &s_lds[p];                     // stride C_DIM*HWT over d
    float m = -3.0e38f;
    for (int d = 0; d < D_DIM; ++d)
      m = fmaxf(m, (float)col[d * (C_DIM * HWT)]);
    float sum = 0.0f;
    for (int d = 0; d < D_DIM; ++d) {
      const float e = __expf((float)col[d * (C_DIM * HWT)] - m);
      sum += e;
      col[d * (C_DIM * HWT)] = (__bf16)e;        // unnormalized weight
    }
    inv_lds[p] = 1.0f / sum;                     // normalize in fp32 at use
  }
  __syncthreads();

  // ---------------- pass 3: V projection, apply attention ----------------
  load_A(Wv);
  gemm_pass(bv_lds, true);
}

extern "C" void kernel_launch(void* const* d_in, const int* in_sizes, int n_in,
                              void* d_out, int out_size, void* d_ws, size_t ws_size,
                              hipStream_t stream) {
  (void)in_sizes; (void)n_in; (void)d_ws; (void)ws_size; (void)out_size;
  const float* x_slice = (const float*)d_in[0];
  const float* x_vol   = (const float*)d_in[1];
  const float* Wq      = (const float*)d_in[2];
  const float* bq      = (const float*)d_in[3];
  const float* Wk      = (const float*)d_in[4];
  const float* bk      = (const float*)d_in[5];
  const float* Wv      = (const float*)d_in[6];
  const float* bv      = (const float*)d_in[7];
  float* out = (float*)d_out;

  hipFuncSetAttribute((const void*)xattn_fused,
                      hipFuncAttributeMaxDynamicSharedMemorySize, SMEM_BYTES);
  xattn_fused<<<dim3(HW_DIM / HWT), dim3(256), SMEM_BYTES, stream>>>(
      x_slice, x_vol, Wq, bq, Wk, bk, Wv, bv, out);
}
